// TokenRoutedMLP_34248069218521
// MI455X (gfx1250) — compile-verified, matrix operand order checked by reading, not run on hardware
//
#include <hip/hip_runtime.h>
#include <math.h>

#define NTOK   4096
#define HID    2048
#define IEXP   1024
#define TWOIE  2048
#define NEXP   8
#define VOCAB  32000
#define BM     32
#define KC     64          // K-chunk per LDS stage (2 WMMA k-steps)
#define KP     72          // bf16 row stride for 64-wide K tiles (144B: 16B-aligned, banks spread)
#define XPAD   76          // f32 row stride, X tile (fallback)
#define WPAD   68          // f32 row stride, W tiles (fallback)
#define IPAD   1032        // bf16 row stride, inter (2064B: 16B-aligned, banks spread)

#if __has_builtin(__builtin_amdgcn_global_load_async_to_lds_b128) && \
    __has_builtin(__builtin_amdgcn_s_wait_asynccnt)
#define HAVE_ASYNC 1
#else
#define HAVE_ASYNC 0
#endif

typedef __attribute__((ext_vector_type(16))) __bf16 v16bf;
typedef __attribute__((ext_vector_type(8)))  float  v8f;
typedef __attribute__((__vector_size__(16))) int    v4i;   // pointee type the async builtin expects

__device__ __forceinline__ __bf16 f2bf(float f) {
    union { float f; unsigned u; } v; v.f = f;
    unsigned r = v.u + 0x7FFFu + ((v.u >> 16) & 1u);   // round-to-nearest-even
    return __builtin_bit_cast(__bf16, (unsigned short)(r >> 16));
}

// 16B global->LDS copy: async DMA on gfx1250, VGPR round-trip fallback.
__device__ __forceinline__ void stage16(const void* src, void* dst) {
#if HAVE_ASYNC
    __builtin_amdgcn_global_load_async_to_lds_b128((v4i*)src, (v4i*)dst, 0, 0);
#else
    *(float4*)dst = *(const float4*)src;
#endif
}

#if HAVE_ASYNC
#define WAIT_ASYNC(n) __builtin_amdgcn_s_wait_asynccnt(n)
#else
#define WAIT_ASYNC(n) ((void)0)
#endif

// ISA 16-bit WMMA operand layout: lanes 0-15 hold K 0..7 & 16..23,
// lanes 16-31 hold K 8..15 & 24..31 (pairs packed per VGPR).
__device__ __forceinline__ v16bf frag_bf(const __bf16* p, int lane) {        // contiguous K
    const int kb = (lane >> 4) << 3;
    v16bf f;
#pragma unroll
    for (int j = 0; j < 8; ++j) {
        const int k = kb + ((j & 3) << 1) + ((j >> 2) << 4);
        f[2 * j]     = p[k];
        f[2 * j + 1] = p[k + 1];
    }
    return f;
}

__device__ __forceinline__ v16bf frag_f32(const float* p, int lane) {        // fallback A
    const int kb = (lane >> 4) << 3;
    v16bf f;
#pragma unroll
    for (int j = 0; j < 8; ++j) {
        const int k = kb + ((j & 3) << 1) + ((j >> 2) << 4);
        f[2 * j]     = f2bf(p[k]);
        f[2 * j + 1] = f2bf(p[k + 1]);
    }
    return f;
}

__device__ __forceinline__ v16bf frag_f32_strided(const float* p, int stride, int lane) { // fallback B
    const int kb = (lane >> 4) << 3;
    v16bf f;
#pragma unroll
    for (int j = 0; j < 8; ++j) {
        const int k = kb + ((j & 3) << 1) + ((j >> 2) << 4);
        f[2 * j]     = f2bf(p[k * stride]);
        f[2 * j + 1] = f2bf(p[(k + 1) * stride]);
    }
    return f;
}

// ---------------- Kernel 1: mu router ----------------
__global__ void router_kernel(const float* __restrict__ mu,
                              const float* __restrict__ wr,
                              const int*   __restrict__ token_ids,
                              int* __restrict__ expert_id) {
    const int t    = blockIdx.x * 8 + (threadIdx.x >> 5);
    const int lane = threadIdx.x & 31;
    if (t >= NTOK) return;
    float acc[NEXP];
#pragma unroll
    for (int e = 0; e < NEXP; ++e) acc[e] = 0.f;
    for (int h = lane; h < HID; h += 32) {
        const float m = mu[(size_t)t * HID + h];
#pragma unroll
        for (int e = 0; e < NEXP; ++e) acc[e] += m * wr[e * HID + h];
    }
#pragma unroll
    for (int e = 0; e < NEXP; ++e)
#pragma unroll
        for (int off = 16; off > 0; off >>= 1)
            acc[e] += __shfl_xor(acc[e], off, 32);
    if (lane == 0) {
        int tid = token_ids[t];
        tid = tid < 0 ? 0 : (tid > VOCAB - 1 ? VOCAB - 1 : tid);
        const int base = tid & (NEXP - 1);
        int best = 0;
        float bv = acc[0] + (base == 0 ? 10.f : 0.f);
#pragma unroll
        for (int e = 1; e < NEXP; ++e) {
            const float v = acc[e] + (base == e ? 10.f : 0.f);
            if (v > bv) { bv = v; best = e; }
        }
        expert_id[t] = best;
    }
}

// ---------------- Kernel 2: bucket tokens by expert ----------------
__global__ void zero_counts_kernel(int* __restrict__ counts) {
    if (threadIdx.x < NEXP) counts[threadIdx.x] = 0;
}

__global__ void bucket_kernel(const int* __restrict__ expert_id,
                              int* __restrict__ counts,
                              int* __restrict__ lists) {
    const int t = blockIdx.x * 256 + threadIdx.x;
    if (t >= NTOK) return;
    const int e   = expert_id[t];
    const int pos = atomicAdd(&counts[e], 1);
    lists[e * NTOK + pos] = t;
}

// ---------------- Pre-pass: f32 -> bf16 conversion ----------------
__global__ void convert_x_kernel(const float* __restrict__ in,
                                 __bf16* __restrict__ out, int n4) {
    const int i = blockIdx.x * 256 + threadIdx.x;
    if (i >= n4) return;
    const float4 v = ((const float4*)in)[i];
    __bf16 t[4] = {f2bf(v.x), f2bf(v.y), f2bf(v.z), f2bf(v.w)};
    ((uint2*)out)[i] = *(uint2*)t;
}

// per expert: in[R][C] f32 -> outT[C][R] bf16 ; grid (C/64, R/64, E), 256 thr
__global__ __launch_bounds__(256) void transpose_bf16_kernel(
        const float* __restrict__ in, __bf16* __restrict__ outT, int R, int C) {
    __shared__ alignas(16) __bf16 sT[64][KP];
    const int tid = threadIdx.x;
    const int n0 = blockIdx.x * 64, k0 = blockIdx.y * 64;
    const size_t eIn  = (size_t)blockIdx.z * R * C;
    const size_t eOut = (size_t)blockIdx.z * C * R;
#pragma unroll
    for (int i = 0; i < 4; ++i) {
        const int c  = tid + i * 256;
        const int kl = c >> 4, nl = (c & 15) << 2;
        const float4 v = *(const float4*)(in + eIn + (size_t)(k0 + kl) * C + n0 + nl);
        sT[kl][nl]     = f2bf(v.x);
        sT[kl][nl + 1] = f2bf(v.y);
        sT[kl][nl + 2] = f2bf(v.z);
        sT[kl][nl + 3] = f2bf(v.w);
    }
    __syncthreads();
    const int n = tid >> 2, ks = (tid & 3) << 4;
    __bf16 tmp[16];
#pragma unroll
    for (int j = 0; j < 16; ++j) tmp[j] = sT[ks + j][n];
    __bf16* dst = outT + eOut + (size_t)(n0 + n) * R + k0 + ks;
    *(uint4*)dst       = *(uint4*)tmp;
    *((uint4*)dst + 1) = *((uint4*)tmp + 1);
}

// ---------------- Kernel 3a: fused grouped MoE MLP, bf16 inputs, async double-buffered ----------------
__global__ __launch_bounds__(256) void moe_mlp_bf16_kernel(
        const __bf16* __restrict__ xb,     // [T][H]
        const __bf16* __restrict__ gupT,   // [E][2IE][H]  (n-major, k contiguous)
        const __bf16* __restrict__ downT,  // [E][H][IE]   (n-major, k contiguous)
        const int*   __restrict__ counts,
        const int*   __restrict__ lists,
        float* __restrict__ out) {
    const int blocks_per_e = NTOK / BM;
    const int e  = blockIdx.x / blocks_per_e;
    const int mb = blockIdx.x % blocks_per_e;

    const int cnt   = counts[e];
    const int start = mb * BM;
    if (start >= cnt) return;
    const int nvalid = (cnt - start) < BM ? (cnt - start) : BM;

    const int tid  = threadIdx.x;
    const int lane = tid & 31;
    const int wave = tid >> 5;
    const int mi   = wave >> 2;
    const int ni   = wave & 3;

    __shared__ alignas(16) __bf16 sXb[2][BM][KP];
    __shared__ alignas(16) __bf16 sWg[2][KC][KP];
    __shared__ alignas(16) __bf16 sWu[2][KC][KP];
    __shared__ alignas(16) __bf16 sI [BM][IPAD];
    __shared__ int           sTok [BM];
    __shared__ const __bf16* sXrow[BM];

    if (tid < BM) {
        const int tk = (tid < nvalid) ? lists[e * NTOK + start + tid] : 0;
        sTok[tid]  = tk;
        sXrow[tid] = xb + (size_t)tk * HID;      // invalid rows read token 0 (never stored)
    }
    __syncthreads();

    const __bf16* Wg = gupT  + (size_t)e * TWOIE * HID;
    const __bf16* Wd = downT + (size_t)e * HID * IEXP;
    const int mrow = mi * 16 + (lane & 15);
    const int ncl  = ni * 16 + (lane & 15);

    // per-thread staging coordinates (16B = 8 bf16 chunks)
    const int xr = tid >> 3,      xk = (tid & 7) << 3;           // X: 32r x 64k = 256 chunks
    const int wn0 = tid >> 3,     wk0 = (tid & 7) << 3;          // W: 64n x 64k = 512 chunks
    const int wn1 = (tid + 256) >> 3, wk1 = ((tid + 256) & 7) << 3;

    // ---------- Phase A: inter = silu(x*Wg) * (x*Wu) ----------
    for (int no = 0; no < IEXP / 64; ++no) {
        const int ncol = no * 64;
        v8f accg = {0.f, 0.f, 0.f, 0.f, 0.f, 0.f, 0.f, 0.f};
        v8f accu = {0.f, 0.f, 0.f, 0.f, 0.f, 0.f, 0.f, 0.f};

        auto stageA = [&](int k0, int b) {
            stage16(sXrow[xr] + k0 + xk, &sXb[b][xr][xk]);
            const size_t o0 = (size_t)(ncol + wn0) * HID + k0 + wk0;
            const size_t o1 = (size_t)(ncol + wn1) * HID + k0 + wk1;
            stage16(Wg + o0,                        &sWg[b][wn0][wk0]);
            stage16(Wg + o0 + (size_t)IEXP * HID,   &sWu[b][wn0][wk0]);
            stage16(Wg + o1,                        &sWg[b][wn1][wk1]);
            stage16(Wg + o1 + (size_t)IEXP * HID,   &sWu[b][wn1][wk1]);
        };

        stageA(0, 0);
        int ib = 0;
#pragma unroll 1
        for (int k0 = 0; k0 < HID; k0 += KC, ib ^= 1) {
            if (k0 + KC < HID) { stageA(k0 + KC, ib ^ 1); WAIT_ASYNC(5); }
            else               { WAIT_ASYNC(0); }
            __syncthreads();
#pragma unroll
            for (int kk = 0; kk < KC; kk += 32) {
                const v16bf a  = frag_bf(&sXb[ib][mrow][kk], lane);
                const v16bf bg = frag_bf(&sWg[ib][ncl][kk], lane);
                const v16bf bu = frag_bf(&sWu[ib][ncl][kk], lane);
                accg = __builtin_amdgcn_wmma_f32_16x16x32_bf16(false, a, false, bg,
                                                               (short)0, accg, false, false);
                accu = __builtin_amdgcn_wmma_f32_16x16x32_bf16(false, a, false, bu,
                                                               (short)0, accu, false, false);
            }
            __syncthreads();
        }
        // silu(g)*u -> inter (C layout: VGPR r -> M=r / r+8; lane%16 -> N)
#pragma unroll
        for (int r = 0; r < 8; ++r) {
            const int row = mi * 16 + r + ((lane >= 16) ? 8 : 0);
            const float g = accg[r];
            const float u = accu[r];
            sI[row][ncol + ni * 16 + (lane & 15)] = f2bf((g / (1.f + __expf(-g))) * u);
        }
    }
    __syncthreads();

    // ---------- Phase B: out = inter @ Wd ----------
    for (int no = 0; no < HID / 64; ++no) {
        const int ncol = no * 64;
        v8f acc = {0.f, 0.f, 0.f, 0.f, 0.f, 0.f, 0.f, 0.f};

        auto stageB = [&](int k0, int b) {
            stage16(Wd + (size_t)(ncol + wn0) * IEXP + k0 + wk0, &sWg[b][wn0][wk0]);
            stage16(Wd + (size_t)(ncol + wn1) * IEXP + k0 + wk1, &sWg[b][wn1][wk1]);
        };

        stageB(0, 0);
        int ib = 0;
#pragma unroll 1
        for (int k0 = 0; k0 < IEXP; k0 += KC, ib ^= 1) {
            if (k0 + KC < IEXP) { stageB(k0 + KC, ib ^ 1); WAIT_ASYNC(2); }
            else                { WAIT_ASYNC(0); }
            __syncthreads();
#pragma unroll
            for (int kk = 0; kk < KC; kk += 32) {
                const v16bf a = frag_bf(&sI[mrow][k0 + kk], lane);
                const v16bf b = frag_bf(&sWg[ib][ncl][kk], lane);
                acc = __builtin_amdgcn_wmma_f32_16x16x32_bf16(false, a, false, b,
                                                              (short)0, acc, false, false);
            }
            __syncthreads();
        }
#pragma unroll
        for (int r = 0; r < 8; ++r) {
            const int rloc = mi * 16 + r + ((lane >= 16) ? 8 : 0);
            if (rloc < nvalid)
                out[(size_t)sTok[rloc] * HID + ncol + ni * 16 + (lane & 15)] = acc[r];
        }
    }
}

// ---------------- Kernel 3b: fallback (f32 staging, inline convert) ----------------
__global__ __launch_bounds__(256) void moe_mlp_f32_kernel(
        const float* __restrict__ x,
        const float* __restrict__ gup,
        const float* __restrict__ down,
        const int*   __restrict__ counts,
        const int*   __restrict__ lists,
        float* __restrict__ out) {
    const int blocks_per_e = NTOK / BM;
    const int e  = blockIdx.x / blocks_per_e;
    const int mb = blockIdx.x % blocks_per_e;
    const int cnt   = counts[e];
    const int start = mb * BM;
    if (start >= cnt) return;
    const int nvalid = (cnt - start) < BM ? (cnt - start) : BM;

    const int tid  = threadIdx.x;
    const int lane = tid & 31;
    const int wave = tid >> 5;
    const int mi   = wave >> 2;
    const int ni   = wave & 3;

    __shared__ alignas(16) float  sXf [BM][XPAD];
    __shared__ alignas(16) float  sWgf[KC][WPAD];
    __shared__ alignas(16) float  sWuf[KC][WPAD];
    __shared__ alignas(16) __bf16 sI  [BM][IPAD];
    __shared__ int          sTok [BM];
    __shared__ const float* sXrow[BM];

    if (tid < BM) {
        const int tk = (tid < nvalid) ? lists[e * NTOK + start + tid] : 0;
        sTok[tid]  = tk;
        sXrow[tid] = x + (size_t)tk * HID;
    }
    __syncthreads();

    const float* Wgu = gup  + (size_t)e * HID * TWOIE;
    const float* Wd  = down + (size_t)e * IEXP * HID;
    const int mrow = mi * 16 + (lane & 15);
    const int ncl  = ni * 16 + (lane & 15);

    for (int no = 0; no < IEXP / 64; ++no) {
        const int ncol = no * 64;
        v8f accg = {0.f, 0.f, 0.f, 0.f, 0.f, 0.f, 0.f, 0.f};
        v8f accu = {0.f, 0.f, 0.f, 0.f, 0.f, 0.f, 0.f, 0.f};
#pragma unroll 1
        for (int k0 = 0; k0 < HID; k0 += KC) {
#pragma unroll
            for (int i = 0; i < 2; ++i) {
                const int c = tid + i * 256;
                const int r = c >> 4, k = (c & 15) << 2;
                stage16(sXrow[r] + k0 + k, &sXf[r][k]);
            }
#pragma unroll
            for (int i = 0; i < 4; ++i) {
                const int c = tid + i * 256;
                const int K = c >> 4, n = (c & 15) << 2;
                const size_t off = (size_t)(k0 + K) * TWOIE + ncol + n;
                stage16(Wgu + off,        &sWgf[K][n]);
                stage16(Wgu + off + IEXP, &sWuf[K][n]);
            }
            WAIT_ASYNC(0);
            __syncthreads();
#pragma unroll
            for (int kk = 0; kk < KC; kk += 32) {
                const v16bf a  = frag_f32(&sXf[mrow][kk], lane);
                const v16bf bg = frag_f32_strided(&sWgf[kk][ncl], WPAD, lane);
                const v16bf bu = frag_f32_strided(&sWuf[kk][ncl], WPAD, lane);
                accg = __builtin_amdgcn_wmma_f32_16x16x32_bf16(false, a, false, bg,
                                                               (short)0, accg, false, false);
                accu = __builtin_amdgcn_wmma_f32_16x16x32_bf16(false, a, false, bu,
                                                               (short)0, accu, false, false);
            }
            __syncthreads();
        }
#pragma unroll
        for (int r = 0; r < 8; ++r) {
            const int row = mi * 16 + r + ((lane >= 16) ? 8 : 0);
            const float g = accg[r];
            const float u = accu[r];
            sI[row][ncol + ni * 16 + (lane & 15)] = f2bf((g / (1.f + __expf(-g))) * u);
        }
    }
    __syncthreads();

    for (int no = 0; no < HID / 64; ++no) {
        const int ncol = no * 64;
        v8f acc = {0.f, 0.f, 0.f, 0.f, 0.f, 0.f, 0.f, 0.f};
#pragma unroll 1
        for (int k0 = 0; k0 < IEXP; k0 += KC) {
#pragma unroll
            for (int i = 0; i < 4; ++i) {
                const int c = tid + i * 256;
                const int K = c >> 4, n = (c & 15) << 2;
                stage16(Wd + (size_t)(k0 + K) * HID + ncol + n, &sWgf[K][n]);
            }
            WAIT_ASYNC(0);
            __syncthreads();
#pragma unroll
            for (int kk = 0; kk < KC; kk += 32) {
                const v16bf a = frag_bf(&sI[mrow][k0 + kk], lane);
                const v16bf b = frag_f32_strided(&sWgf[kk][ncl], WPAD, lane);
                acc = __builtin_amdgcn_wmma_f32_16x16x32_bf16(false, a, false, b,
                                                              (short)0, acc, false, false);
            }
            __syncthreads();
        }
#pragma unroll
        for (int r = 0; r < 8; ++r) {
            const int rloc = mi * 16 + r + ((lane >= 16) ? 8 : 0);
            if (rloc < nvalid)
                out[(size_t)sTok[rloc] * HID + ncol + ni * 16 + (lane & 15)] = acc[r];
        }
    }
}

extern "C" void kernel_launch(void* const* d_in, const int* in_sizes, int n_in,
                              void* d_out, int out_size, void* d_ws, size_t ws_size,
                              hipStream_t stream) {
    const float* x    = (const float*)d_in[0];
    const int*   tids = (const int*)  d_in[1];
    const float* mu   = (const float*)d_in[2];
    const float* gup  = (const float*)d_in[3];
    const float* down = (const float*)d_in[4];
    const float* wr   = (const float*)d_in[5];
    float* out = (float*)d_out;

    int* expert_id = (int*)d_ws;                 // NTOK ints
    int* counts    = expert_id + NTOK;           // NEXP ints
    int* lists     = counts + NEXP;              // NEXP*NTOK ints

    router_kernel<<<NTOK / 8, 256, 0, stream>>>(mu, wr, tids, expert_id);
    zero_counts_kernel<<<1, 32, 0, stream>>>(counts);
    bucket_kernel<<<NTOK / 256, 256, 0, stream>>>(expert_id, counts, lists);

    const size_t metaB = ((size_t)(NTOK + NEXP + NEXP * NTOK) * 4 + 255) & ~(size_t)255;
    const size_t nXb   = (size_t)NTOK * HID;
    const size_t nGup  = (size_t)NEXP * TWOIE * HID;
    const size_t nDown = (size_t)NEXP * HID * IEXP;
    const size_t need  = metaB + 2 * (nXb + nGup + nDown);

    if (ws_size >= need) {
        __bf16* xbf   = (__bf16*)((char*)d_ws + metaB);
        __bf16* gupT  = xbf + nXb;
        __bf16* downT = gupT + nGup;
        convert_x_kernel<<<(int)(nXb / 4 / 256), 256, 0, stream>>>(x, xbf, (int)(nXb / 4));
        transpose_bf16_kernel<<<dim3(TWOIE / 64, HID / 64, NEXP), 256, 0, stream>>>(gup, gupT, HID, TWOIE);
        transpose_bf16_kernel<<<dim3(HID / 64, IEXP / 64, NEXP), 256, 0, stream>>>(down, downT, IEXP, HID);
        moe_mlp_bf16_kernel<<<NEXP * (NTOK / BM), 256, 0, stream>>>(xbf, gupT, downT,
                                                                    counts, lists, out);
    } else {
        moe_mlp_f32_kernel<<<NEXP * (NTOK / BM), 256, 0, stream>>>(x, gup, down,
                                                                   counts, lists, out);
    }
}
